// DivergentAttention_35459249995777
// MI455X (gfx1250) — matrix operand order repaired and checked by reading, not compiled, not run on hardware
//
#include <hip/hip_runtime.h>
#include <hip/hip_bf16.h>

typedef unsigned short u16;
typedef unsigned int   u32;
typedef unsigned long long u64;
typedef __attribute__((ext_vector_type(16))) __bf16 v16bf;
typedef __attribute__((ext_vector_type(8)))  float  v8f;

#define S_LEN 2048
#define D_MOD 1024
#define NH    16
#define HD    64
#define BATCH 2
#define MROWS (BATCH * S_LEN)          // 4096
#define LOG_FOCUS 0.4700036292457356f  // ln(1.6)
#define NEG_INF  -3.0e38f

union FragBF {
  v16bf v;
  u16   u[16];
  uint4 q[2];
};

__device__ __forceinline__ u16 f2bf(float x) {
  u32 u = __builtin_bit_cast(u32, x);
  u32 r = u + 0x7FFFu + ((u >> 16) & 1u);   // round-to-nearest-even
  return (u16)(r >> 16);
}

// A-fragment (16x32, bf16, row-major source with leading dim ld):
//   lane<16 : row=lane,    elems e -> K=kb+e
//   lane>=16: row=lane-16, elems e -> K=kb+16+e
__device__ __forceinline__ v16bf load_a_frag(const u16* src, int ld, int m0, int kb, int lane) {
  int m  = m0 + (lane & 15);
  int kq = kb + ((lane >> 4) << 4);
  FragBF f;
  f.q[0] = *(const uint4*)(src + m * ld + kq);
  f.q[1] = *(const uint4*)(src + m * ld + kq + 8);
  return f.v;
}

// B-fragment (32x16, bf16). Source laid out K-contiguous per column: src[n*ld + k].
//   lane<16 : n=nb+lane,    elems e -> K=kb+e
//   lane>=16: n=nb+lane-16, elems e -> K=kb+16+e
__device__ __forceinline__ v16bf load_b_frag(const u16* src, int ld, int nb, int kb, int lane) {
  int n = nb + (lane & 15);
  int k = kb + ((lane >> 4) << 4);
  FragBF f;
  const uint4* p = (const uint4*)(src + n * ld + k);
  f.q[0] = p[0];
  f.q[1] = p[1];
  return f.v;
}

__device__ __forceinline__ v8f wmma_bf16(v16bf a, v16bf b, v8f c) {
  return __builtin_amdgcn_wmma_f32_16x16x32_bf16(false, a, false, b, (short)0, c, false, false);
}

// Async-copy 32 bytes global -> LDS (two B128 transfers, ASYNCcnt-tracked).
// INST_OFFSET applies to both the LDS and global addresses (ISA 15.18 pseudocode),
// so one lds-addr VGPR + one global-addr VGPR pair covers both halves.
__device__ __forceinline__ void stage32_async(const u16* lptr, const u16* gptr) {
  u32 l = (u32)(size_t)lptr;                 // LDS address is the low 32 bits of the pointer
  u64 g = (u64)(size_t)gptr;
  asm volatile("global_load_async_to_lds_b128 %0, %1, off\n\t"
               "global_load_async_to_lds_b128 %0, %1, off offset:16"
               :: "v"(l), "v"(g) : "memory");
}

__device__ __forceinline__ void wait_async0() {
  asm volatile("s_wait_asynccnt 0" ::: "memory");
}

// ---------------------------------------------------------------- conversions
__global__ void da_cvt_bf16(const float* __restrict__ in, u16* __restrict__ out, int n) {
  int i = blockIdx.x * blockDim.x + threadIdx.x;
  if (i < n) out[i] = f2bf(in[i]);
}

// in: [K][N] f32  ->  out: [N][K] bf16
__global__ void da_cvt_transpose(const float* __restrict__ in, u16* __restrict__ out, int K, int N) {
  int i = blockIdx.x * blockDim.x + threadIdx.x;
  if (i < K * N) {
    int k = i / N, n = i % N;
    out[n * K + k] = f2bf(in[i]);
  }
}

// ---------------------------------------------------------------- GEMM core
// Block: 256 threads = 8 waves; block tile 128M x 64N. B tile (64N x 64K bf16,
// 8KB) staged to LDS with async-to-LDS, double buffered, one barrier per step.
// The top-of-iteration barrier alone is sufficient: each wave waits dscnt
// before consuming fragments, so passing the barrier implies the previous
// buffer's reads are complete before it is overwritten.
__device__ __forceinline__ void gemm_acc(
    const u16* __restrict__ A, const u16* __restrict__ Wt,
    u16* btile /* [2][64*64] */, int m0, int nb, int tid, int lane, v8f acc[4]) {
  const int cn = tid >> 2;            // 0..63 : B column
  const int ck = (tid & 3) * 16;      // 0/16/32/48 : K offset
  const u16* gsrc = Wt + (size_t)(nb + cn) * D_MOD + ck;
  u16* l0 = btile + cn * 64 + ck;
  u16* l1 = l0 + 64 * 64;

  stage32_async(l0, gsrc);            // prologue: tile kk=0 -> buf0

  const int arow = m0 + (lane & 15);
  int it = 0;
  for (int kk = 0; kk < D_MOD; kk += 64, ++it) {
    wait_async0();
    __syncthreads();
    const u16* bufc = btile + (it & 1) * (64 * 64);
    if (kk + 64 < D_MOD) {            // overlap next copy with this tile's math
      stage32_async((it & 1) ? l0 : l1, gsrc + kk + 64);
      __builtin_prefetch(A + arow * D_MOD + kk + 64, 0, 1);
    }
#pragma unroll
    for (int ks = 0; ks < 64; ks += 32) {
      v16bf a  = load_a_frag(A, D_MOD, m0, kk + ks, lane);
      v16bf b0 = load_b_frag(bufc, 64, 0,  ks, lane);
      v16bf b1 = load_b_frag(bufc, 64, 16, ks, lane);
      v16bf b2 = load_b_frag(bufc, 64, 32, ks, lane);
      v16bf b3 = load_b_frag(bufc, 64, 48, ks, lane);
      acc[0] = wmma_bf16(a, b0, acc[0]);
      acc[1] = wmma_bf16(a, b1, acc[1]);
      acc[2] = wmma_bf16(a, b2, acc[2]);
      acc[3] = wmma_bf16(a, b3, acc[3]);
    }
  }
}

// ---------------------------------------------------------------- QKV GEMM
// Xb:[4096][1024] bf16; Wt:[3072][1024] bf16 ([N][K], K contiguous)
// Writes q,k in [B][H][S][HD] bf16 (q pre-scaled by 1/8), v transposed [B][H][HD][S].
__global__ __launch_bounds__(256) void da_qkv_gemm(
    const u16* __restrict__ Xb, const u16* __restrict__ Wt,
    const float* __restrict__ bias,
    u16* __restrict__ qb, u16* __restrict__ kbuf, u16* __restrict__ vT) {
  __shared__ __align__(16) u16 btile[2][64 * 64];   // 2 x 8KB

  const int tid  = threadIdx.x;
  const int lane = tid & 31;
  const int wave = tid >> 5;
  const int m0 = (blockIdx.y * 8 + wave) * 16;
  const int nb = blockIdx.x * 64;

  v8f acc[4];
#pragma unroll
  for (int t = 0; t < 4; ++t) acc[t] = (v8f){};

  gemm_acc(Xb, Wt, &btile[0][0], m0, nb, tid, lane, acc);

#pragma unroll
  for (int t = 0; t < 4; ++t) {
    const int col = nb + 16 * t + (lane & 15);  // 0..3071
    const int part = col >> 10;                 // 0=q 1=k 2=v
    const int d   = col & (D_MOD - 1);
    const int h   = d >> 6;
    const int hdp = d & 63;
    const float bv = bias[col];
#pragma unroll
    for (int r = 0; r < 8; ++r) {
      const int row = m0 + r + ((lane >> 4) << 3);
      const int bidx = row >> 11;       // batch
      const int s    = row & (S_LEN - 1);
      float v = acc[t][r] + bv;
      if (part == 0) {
        v *= 0.125f;  // fold 1/sqrt(hd) into Q
        qb[(((bidx * NH + h) * S_LEN) + s) * HD + hdp] = f2bf(v);
      } else if (part == 1) {
        kbuf[(((bidx * NH + h) * S_LEN) + s) * HD + hdp] = f2bf(v);
      } else {
        vT[(((bidx * NH + h) * HD) + hdp) * S_LEN + s] = f2bf(v);
      }
    }
  }
}

// ---------------------------------------------------------------- attention
__device__ __forceinline__ float region_bias(int h, int k) {
  // first_end = S/3 = 682, second_end = 2S/3 = 1365 (matches python floor div)
  if (h == 0) return (k < 682) ? LOG_FOCUS : 0.0f;
  if (h == 1) return (k >= 682 && k < 1365) ? LOG_FOCUS : 0.0f;
  if (h == 2) return (k >= 1365) ? LOG_FOCUS : 0.0f;
  return 0.0f;
}

__global__ __launch_bounds__(128) void da_attn(
    const u16* __restrict__ qb, const u16* __restrict__ kbuf,
    const u16* __restrict__ vT, u16* __restrict__ aout) {
  __shared__ __align__(16) u16 plds[4][16 * 32];

  const int lane = threadIdx.x & 31;
  const int wave = threadIdx.x >> 5;
  const int id = blockIdx.x * 4 + wave;
  const int qi = id & 127;        // query tile within (b,h)
  const int bh = id >> 7;
  const int h  = bh & (NH - 1);
  const int b  = bh >> 4;
  const int q0 = qi * 16;

  const u16* Q  = qb   + ((size_t)(b * NH + h) * S_LEN) * HD;
  const u16* Kh = kbuf + ((size_t)(b * NH + h) * S_LEN) * HD;
  const u16* Vh = vT   + ((size_t)(b * NH + h) * HD) * S_LEN;

  const v16bf qf0 = load_a_frag(Q, HD, q0, 0, lane);
  const v16bf qf1 = load_a_frag(Q, HD, q0, 32, lane);

  float rm[8], rl[8];
#pragma unroll
  for (int r = 0; r < 8; ++r) { rm[r] = NEG_INF; rl[r] = 0.0f; }
  v8f o[4];
#pragma unroll
  for (int t = 0; t < 4; ++t) o[t] = (v8f){};

  const int rowbase = q0 + ((lane >> 4) << 3);
  const int nkp = (q0 + 16 + 31) >> 5;  // number of 32-key pairs (causal)

  for (int kp = 0; kp < nkp; ++kp) {
    const int kb0 = kp * 32;

    // ---- scores: two 16x16 tiles over hd=64 (two K=32 steps each)
    v8f s0 = (v8f){}, s1 = (v8f){};
    s0 = wmma_bf16(qf0, load_b_frag(Kh, HD, kb0, 0, lane), s0);
    s0 = wmma_bf16(qf1, load_b_frag(Kh, HD, kb0, 32, lane), s0);
    s1 = wmma_bf16(qf0, load_b_frag(Kh, HD, kb0 + 16, 0, lane), s1);
    s1 = wmma_bf16(qf1, load_b_frag(Kh, HD, kb0 + 16, 32, lane), s1);

    const int c0 = kb0 + (lane & 15);
    const int c1 = c0 + 16;
    const float bias0 = region_bias(h, c0);
    const float bias1 = region_bias(h, c1);

#pragma unroll
    for (int r = 0; r < 8; ++r) {
      const int qrow = rowbase + r;
      float v0 = s0[r] + bias0; if (c0 > qrow) v0 = NEG_INF;
      float v1 = s1[r] + bias1; if (c1 > qrow) v1 = NEG_INF;

      // row max across the 16 lanes holding this row (xor masks stay in half)
      float tm = fmaxf(v0, v1);
      tm = fmaxf(tm, __shfl_xor(tm, 1));
      tm = fmaxf(tm, __shfl_xor(tm, 2));
      tm = fmaxf(tm, __shfl_xor(tm, 4));
      tm = fmaxf(tm, __shfl_xor(tm, 8));

      const float mnew  = fmaxf(rm[r], tm);
      const float alpha = __expf(rm[r] - mnew);
      rm[r] = mnew;

      const float p0 = __expf(v0 - mnew);
      const float p1 = __expf(v1 - mnew);
      float ps = p0 + p1;
      ps += __shfl_xor(ps, 1);
      ps += __shfl_xor(ps, 2);
      ps += __shfl_xor(ps, 4);
      ps += __shfl_xor(ps, 8);
      rl[r] = rl[r] * alpha + ps;

      const int trow = r + ((lane >> 4) << 3);
      plds[wave][trow * 32 + (lane & 15)]      = f2bf(p0);
      plds[wave][trow * 32 + 16 + (lane & 15)] = f2bf(p1);

#pragma unroll
      for (int t = 0; t < 4; ++t) o[t][r] *= alpha;
    }

    // ---- P (LDS, C-layout -> A-layout) x V^T
    {
      const int m  = lane & 15;
      const int kq = (lane >> 4) << 4;
      FragBF pf;
      pf.q[0] = *(const uint4*)(&plds[wave][m * 32 + kq]);
      pf.q[1] = *(const uint4*)(&plds[wave][m * 32 + 8 + kq]);
#pragma unroll
      for (int t = 0; t < 4; ++t) {
        v16bf vb = load_b_frag(Vh, S_LEN, 16 * t, kb0, lane);
        o[t] = wmma_bf16(pf.v, vb, o[t]);
      }
    }
  }

  // ---- write normalized output, [B][S][D] bf16 (head-concat)
#pragma unroll
  for (int t = 0; t < 4; ++t) {
    const int hdc = 16 * t + (lane & 15);
#pragma unroll
    for (int r = 0; r < 8; ++r) {
      const int row = q0 + r + ((lane >> 4) << 3);
      const float val = o[t][r] / rl[r];
      aout[((size_t)(b * S_LEN + row)) * D_MOD + h * HD + hdc] = f2bf(val);
    }
  }
}

// ---------------------------------------------------------------- output proj
// Ab:[4096][1024] bf16; Wt:[1024][1024] bf16 ([N][K]); out f32 [4096][1024]
__global__ __launch_bounds__(256) void da_proj_gemm(
    const u16* __restrict__ Ab, const u16* __restrict__ Wt,
    const float* __restrict__ bias, float* __restrict__ out) {
  __shared__ __align__(16) u16 btile[2][64 * 64];

  const int tid  = threadIdx.x;
  const int lane = tid & 31;
  const int wave = tid >> 5;
  const int m0 = (blockIdx.y * 8 + wave) * 16;
  const int nb = blockIdx.x * 64;

  v8f acc[4];
#pragma unroll
  for (int t = 0; t < 4; ++t) acc[t] = (v8f){};

  gemm_acc(Ab, Wt, &btile[0][0], m0, nb, tid, lane, acc);

#pragma unroll
  for (int t = 0; t < 4; ++t) {
    const int col = nb + 16 * t + (lane & 15);
    const float bv = bias[col];
#pragma unroll
    for (int r = 0; r < 8; ++r) {
      const int row = m0 + r + ((lane >> 4) << 3);
      out[(size_t)row * D_MOD + col] = acc[t][r] + bv;
    }
  }
}

// ---------------------------------------------------------------- launch
extern "C" void kernel_launch(void* const* d_in, const int* in_sizes, int n_in,
                              void* d_out, int out_size, void* d_ws, size_t ws_size,
                              hipStream_t stream) {
  const float* hs  = (const float*)d_in[0];  // [2,2048,1024]
  const float* caw = (const float*)d_in[1];  // [1024,3072]
  const float* cab = (const float*)d_in[2];  // [3072]
  const float* cpw = (const float*)d_in[3];  // [1024,1024]
  const float* cpb = (const float*)d_in[4];  // [1024]
  float* out = (float*)d_out;

  // workspace layout (bf16 elements)
  u16* Xb = (u16*)d_ws;                              // 4096*1024
  u16* Wa = Xb + (size_t)MROWS * D_MOD;              // 3072*1024
  u16* Wp = Wa + (size_t)3 * D_MOD * D_MOD;          // 1024*1024
  u16* qb = Wp + (size_t)D_MOD * D_MOD;              // 2*16*2048*64
  u16* kb = qb + (size_t)BATCH * NH * S_LEN * HD;
  u16* vT = kb + (size_t)BATCH * NH * S_LEN * HD;
  u16* ao = vT + (size_t)BATCH * NH * S_LEN * HD;    // 4096*1024

  const int nX = MROWS * D_MOD;
  da_cvt_bf16<<<nX / 256, 256, 0, stream>>>(hs, Xb, nX);
  da_cvt_transpose<<<(D_MOD * 3 * D_MOD) / 256, 256, 0, stream>>>(caw, Wa, D_MOD, 3 * D_MOD);
  da_cvt_transpose<<<(D_MOD * D_MOD) / 256, 256, 0, stream>>>(cpw, Wp, D_MOD, D_MOD);

  // QKV: M=4096, N=3072; block tile 128M x 64N
  da_qkv_gemm<<<dim3(3 * D_MOD / 64, MROWS / 128), 256, 0, stream>>>(Xb, Wa, cab, qb, kb, vT);

  // attention: 2*16*128 q-tiles / 4 waves per block
  da_attn<<<(BATCH * NH * (S_LEN / 16)) / 4, 128, 0, stream>>>(qb, kb, vT, ao);

  // projection: M=4096, N=1024; block tile 128M x 64N
  da_proj_gemm<<<dim3(D_MOD / 64, MROWS / 128), 256, 0, stream>>>(ao, Wp, cpb, out);
}
